// TransformerProjectGOPoincareBall_89051851915573
// MI455X (gfx1250) — matrix operand order repaired and checked by reading, not compiled, not run on hardware
//
#include <hip/hip_runtime.h>

typedef __attribute__((ext_vector_type(16))) __bf16 v16bf;
typedef __attribute__((ext_vector_type(8)))  __bf16 v8bf;
typedef __attribute__((ext_vector_type(8)))  float  v8f;
typedef __attribute__((ext_vector_type(4)))  int    v4i;

#define B_SZ   32
#define L0_SZ  2048
#define CIN    1024
#define CMID   32
#define L2_SZ  514
#define L3_SZ  130
#define L4_SZ  34
#define NPTS   (B_SZ * L4_SZ)      // 1088
#define DBALL  200
#define DPAD   224                 // 7 chunks of 32
#define NGO    10000
#define NGOPAD 10112               // 79 * 128
#define SSTRIDE 10112

// conv1 LDS staging geometry: 134 rows (128 outputs + 6 halo), 32-ch blocks,
// padded row stride 36 floats => 16 consecutive rows hit distinct bank groups.
#define NROWS   134
#define LSTRIDE 36

#if defined(__has_builtin)
#if __has_builtin(__builtin_amdgcn_global_load_async_to_lds_b128) && \
    __has_builtin(__builtin_amdgcn_s_wait_asynccnt)
#define USE_ASYNC 1
#endif
#endif
#ifndef USE_ASYNC
#define USE_ASYNC 0
#endif

#if USE_ASYNC
typedef __attribute__((address_space(1))) v4i ga_v4i;   // global int4*
typedef __attribute__((address_space(3))) v4i la_v4i;   // LDS int4*
#endif

__device__ __forceinline__ void async_fence_lds() {
#if USE_ASYNC
    __builtin_amdgcn_s_wait_asynccnt(0);
#endif
}

// ---------------------------------------------------------------------------
// Pack conv1 weights [32,1024,7] f32 into per-lane B-fragment bf16 layout:
// wpack[k][chunk][nf][lane][i],  N = nf*16 + (lane&15),  K = chunk*32 + (i&7) + 8*(lane>>4) + 16*(i>>3)
// ---------------------------------------------------------------------------
__global__ void k_pack_w(const float* __restrict__ w, __bf16* __restrict__ wpack) {
    int idx = blockIdx.x * 256 + threadIdx.x;          // < 7*32*2*32*16 = 229376
    int i     = idx & 15;
    int lane  = (idx >> 4) & 31;
    int nf    = (idx >> 9) & 1;
    int chunk = (idx >> 10) & 31;
    int k     = idx >> 15;
    int co = nf * 16 + (lane & 15);
    int ci = chunk * 32 + (i & 7) + 8 * (lane >> 4) + 16 * (i >> 3);
    wpack[idx] = (__bf16)w[((size_t)co * CIN + ci) * 7 + k];
}

// ---------------------------------------------------------------------------
// Convert go_emb [10000,200] f32 -> bf16 padded [10112,224]; per-row |v|^2 clipped.
// ---------------------------------------------------------------------------
__global__ void k_go(const float* __restrict__ go, __bf16* __restrict__ gbf,
                     float* __restrict__ v2v) {
    __shared__ float sred[256];
    int tid = threadIdx.x, r = blockIdx.x;
    float v = 0.f;
    if (r < NGO && tid < DBALL) v = go[(size_t)r * DBALL + tid];
    if (tid < DPAD) gbf[(size_t)r * DPAD + tid] = (__bf16)v;
    sred[tid] = v * v; __syncthreads();
    for (int s = 128; s > 0; s >>= 1) { if (tid < s) sred[tid] += sred[tid + s]; __syncthreads(); }
    if (tid == 0) v2v[r] = fminf(fmaxf(sred[0], 0.f), 1.f - 1e-6f);
}

// ---------------------------------------------------------------------------
// conv1 as implicit GEMM via WMMA bf16: M=B*L (65536), N=32, K=7*1024.
// Workgroup = 8 waves = one 128-row M-tile. Input slab (134 rows x 32 ch f32)
// staged read-once into LDS (async global->LDS when available), double
// buffered so the next block's DMA overlaps this block's 14 WMMAs.
// ---------------------------------------------------------------------------
__device__ __forceinline__ void stage_block(const float* __restrict__ x, float* buf,
                                            int b, int ll0, int c0, int tid) {
    // rows ll0-3 .. ll0+130, channels c0..c0+31 -> buf[r*LSTRIDE + c]
    for (int e = tid; e < NROWS * 8; e += 256) {
        int r  = e >> 3;
        int c4 = (e & 7) << 2;             // 0,4,...,28
        int gr = ll0 - 3 + r;
        float* dst = buf + r * LSTRIDE + c4;
        if (gr >= 0 && gr < L0_SZ) {
            const float* src = x + ((size_t)b * L0_SZ + gr) * CIN + c0 + c4;
#if USE_ASYNC
            __builtin_amdgcn_global_load_async_to_lds_b128(
                (ga_v4i*)src, (la_v4i*)dst, 0, 0);
#else
            *(float4*)dst = *(const float4*)src;
#endif
        } else {
            float4 z = {0.f, 0.f, 0.f, 0.f};
            *(float4*)dst = z;
        }
    }
}

__device__ __forceinline__ v8f wmma_bf16(v16bf a, v16bf b, v8f c) {
    return __builtin_amdgcn_wmma_f32_16x16x32_bf16(false, a, false, b,
                                                   (short)0, c, false, false);
}

__global__ void __launch_bounds__(256) k_conv1(const float* __restrict__ x,
                                               const __bf16* __restrict__ wpack,
                                               const float* __restrict__ bias,
                                               float* __restrict__ y1) {
    __shared__ float sbuf[2][NROWS * LSTRIDE];
    const int tid  = threadIdx.x;
    const int lane = tid & 31;
    const int wave = tid >> 5;
    const int mbase = blockIdx.x * 128;
    const int b   = mbase >> 11;          // / 2048
    const int ll0 = mbase & 2047;
    const int hi = lane >> 4;
    const int ms = lane & 15;

    v8f acc0 = {}; v8f acc1 = {};

    stage_block(x, sbuf[0], b, ll0, 0, tid);
    async_fence_lds();
    __syncthreads();

    for (int blk = 0; blk < 32; ++blk) {
        if (blk + 1 < 32)
            stage_block(x, sbuf[(blk + 1) & 1], b, ll0, (blk + 1) * 32, tid);

        const float* buf = sbuf[blk & 1];
#pragma unroll
        for (int k = 0; k < 7; ++k) {
            const int rt = wave * 16 + ms + k;           // A row in LDS slab
            const float* rp = buf + rt * LSTRIDE + 8 * hi;
            float4 f0 = *(const float4*)(rp);
            float4 f1 = *(const float4*)(rp + 4);
            float4 f2 = *(const float4*)(rp + 16);
            float4 f3 = *(const float4*)(rp + 20);
            v16bf a;
            a[0]=(__bf16)f0.x; a[1]=(__bf16)f0.y; a[2]=(__bf16)f0.z; a[3]=(__bf16)f0.w;
            a[4]=(__bf16)f1.x; a[5]=(__bf16)f1.y; a[6]=(__bf16)f1.z; a[7]=(__bf16)f1.w;
            a[8]=(__bf16)f2.x; a[9]=(__bf16)f2.y; a[10]=(__bf16)f2.z; a[11]=(__bf16)f2.w;
            a[12]=(__bf16)f3.x; a[13]=(__bf16)f3.y; a[14]=(__bf16)f3.z; a[15]=(__bf16)f3.w;
            const __bf16* wp = wpack + ((size_t)(k * 32 + blk) * 64 + lane) * 16;
            v16bf b0 = *(const v16bf*)wp;
            v16bf b1 = *(const v16bf*)(wp + 512);       // nf=1
            acc0 = wmma_bf16(a, b0, acc0);
            acc1 = wmma_bf16(a, b1, acc1);
        }
        async_fence_lds();
        __syncthreads();
    }

    const float bi0 = bias[ms], bi1 = bias[16 + ms];
#pragma unroll
    for (int r = 0; r < 8; ++r) {
        int lo = ll0 + wave * 16 + r + 8 * hi;           // C layout: M = r + 8*hi
        float v0 = acc0[r] + bi0; v0 = v0 > 0.f ? v0 : 0.f;
        float v1 = acc1[r] + bi1; v1 = v1 > 0.f ? v1 : 0.f;
        size_t o = ((size_t)b * L0_SZ + lo) * CMID;
        y1[o + ms] = v0;
        y1[o + 16 + ms] = v1;
    }
}

// ---------------------------------------------------------------------------
// Small strided convs (stride 4, pad 6, K=7, 32x32 channels) + ReLU. Scalar.
// ---------------------------------------------------------------------------
__global__ void k_convs(const float* __restrict__ yin, const float* __restrict__ w,
                        const float* __restrict__ bias, float* __restrict__ yout,
                        int Lin, int Lout) {
    int idx = blockIdx.x * 256 + threadIdx.x;
    int co = idx & 31;
    int t  = idx >> 5;
    int l  = t % Lout;
    int b  = t / Lout;
    if (b >= B_SZ) return;
    float acc = bias[co];
    for (int k = 0; k < 7; ++k) {
        int li = l * 4 + k - 6;
        if (li < 0 || li >= Lin) continue;
        const float* xr = yin + ((size_t)b * Lin + li) * 32;
        const float* wr = w + co * 224 + k;            // w[co][ci][k]
#pragma unroll
        for (int ci = 0; ci < 32; ++ci) acc = fmaf(xr[ci], wr[ci * 7], acc);
    }
    acc = acc > 0.f ? acc : 0.f;
    yout[((size_t)b * Lout + l) * 32 + co] = acc;
}

// ---------------------------------------------------------------------------
// expmap0 -> mobius_matvec -> mobius_add, exact f32. One block per point.
// ---------------------------------------------------------------------------
__device__ __forceinline__ float blk_sum(float v, float* sred, int tid) {
    sred[tid] = v; __syncthreads();
    for (int s = 128; s > 0; s >>= 1) { if (tid < s) sred[tid] += sred[tid + s]; __syncthreads(); }
    float r = sred[0]; __syncthreads();
    return r;
}

__global__ void k_hyper(const float* __restrict__ y4, const float* __restrict__ mobw,
                        const float* __restrict__ mobb, __bf16* __restrict__ pbf,
                        float* __restrict__ u2v) {
    __shared__ float sx[32], spx[32], smx[224], smm[224], sb[224], sred[256];
    const int tid = threadIdx.x, m = blockIdx.x;
    if (tid < 32)  sx[tid] = y4[(size_t)m * 32 + tid];
    if (tid < 224) sb[tid] = (tid < DBALL) ? mobb[tid] : 0.f;
    __syncthreads();
    float xn2 = blk_sum(tid < 32 ? sx[tid] * sx[tid] : 0.f, sred, tid);
    float xn = fmaxf(sqrtf(xn2), 1e-15f);
    float sc = tanhf(xn) / xn;
    if (tid < 32) spx[tid] = sc * sx[tid];
    __syncthreads();
    float mx = 0.f;
    if (tid < DBALL) {
        const float* wr = mobw + (size_t)tid * 32;
#pragma unroll
        for (int ci = 0; ci < 32; ++ci) mx = fmaf(wr[ci], spx[ci], mx);
    }
    if (tid < 224) smx[tid] = (tid < DBALL) ? mx : 0.f;
    __syncthreads();
    float pn2  = blk_sum(tid < 32 ? spx[tid] * spx[tid] : 0.f, sred, tid);
    float mxn2 = blk_sum(tid < 224 ? smx[tid] * smx[tid] : 0.f, sred, tid);
    float pn  = fmaxf(sqrtf(pn2), 1e-15f);
    float mxn = fmaxf(sqrtf(mxn2), 1e-15f);
    float z = fminf(pn, 1.f - 1e-5f);
    float atx = 0.5f * logf((1.f + z) / (1.f - z));    // arctanh
    float tt = tanhf(mxn / pn * atx);
    if (tid < 224) smm[tid] = (tid < DBALL) ? tt * smx[tid] / mxn : 0.f;
    __syncthreads();
    float xy = blk_sum(tid < 224 ? smm[tid] * sb[tid] : 0.f, sred, tid);
    float x2 = blk_sum(tid < 224 ? smm[tid] * smm[tid] : 0.f, sred, tid);
    float y2 = blk_sum(tid < 224 ? sb[tid] * sb[tid] : 0.f, sred, tid);
    float den = fmaxf(1.f + 2.f * xy + x2 * y2, 1e-15f);
    float pv = 0.f;
    if (tid < DBALL)
        pv = ((1.f + 2.f * xy + y2) * smm[tid] + (1.f - x2) * sb[tid]) / den;
    float uu = blk_sum(tid < 224 ? pv * pv : 0.f, sred, tid);
    if (tid < 224) pbf[(size_t)m * DPAD + tid] = (__bf16)pv;
    if (tid == 0)  u2v[m] = fminf(fmaxf(uu, 0.f), 1.f - 1e-6f);
}

// ---------------------------------------------------------------------------
// Pairwise Poincare distance via WMMA GEMM, fused arcosh epilogue.
// ---------------------------------------------------------------------------
__global__ void k_dist(const __bf16* __restrict__ pbf, const __bf16* __restrict__ gbf,
                       const float* __restrict__ u2v, const float* __restrict__ v2v,
                       float* __restrict__ S) {
    const int lane = threadIdx.x & 31;
    const int wave = threadIdx.x >> 5;
    const int m0 = blockIdx.x * 16;
    const int n0 = blockIdx.y * 128 + wave * 16;
    const int hi = lane >> 4;
    const int ms = lane & 15;
    const int m = m0 + ms;                 // A row
    const int g = n0 + ms;                 // B column
    const __bf16* prow = pbf + (size_t)m * DPAD;
    const __bf16* grow = gbf + (size_t)g * DPAD;

    v8f acc = {};
#pragma unroll
    for (int chunk = 0; chunk < 7; ++chunk) {
        v16bf a, bb;
        v8bf alo = *(const v8bf*)(prow + chunk * 32 + 8 * hi);
        v8bf ahi = *(const v8bf*)(prow + chunk * 32 + 16 + 8 * hi);
        v8bf blo = *(const v8bf*)(grow + chunk * 32 + 8 * hi);
        v8bf bhi = *(const v8bf*)(grow + chunk * 32 + 16 + 8 * hi);
#pragma unroll
        for (int i = 0; i < 8; ++i) { a[i] = alo[i]; a[8 + i] = ahi[i]; bb[i] = blo[i]; bb[8 + i] = bhi[i]; }
        acc = wmma_bf16(a, bb, acc);
    }

    const float vg = v2v[g];
#pragma unroll
    for (int r = 0; r < 8; ++r) {
        int mm = m0 + r + 8 * hi;          // C layout: M = r + 8*hi
        float uu = u2v[mm];
        float dot = acc[r];
        float sq = uu + vg - 2.f * dot; sq = sq > 0.f ? sq : 0.f;
        float xx = 1.f + 2.f * sq / ((1.f - uu) * (1.f - vg));
        xx = fmaxf(xx, 1.f + 1e-7f);
        float d = logf(xx + sqrtf(xx * xx - 1.f));     // arcosh
        S[(size_t)mm * SSTRIDE + g] = -d;
    }
}

// ---------------------------------------------------------------------------
// Per-row softmax stats over 10000 GO terms: rowlog = rowmax + log(sum exp).
// ---------------------------------------------------------------------------
__global__ void k_rowstats(const float* __restrict__ S, float* __restrict__ rowlog) {
    __shared__ float sred[256];
    const int tid = threadIdx.x, m = blockIdx.x;
    const float* row = S + (size_t)m * SSTRIDE;
    float mx = -3.4e38f;
    for (int gg = tid; gg < NGO; gg += 256) mx = fmaxf(mx, row[gg]);
    sred[tid] = mx; __syncthreads();
    for (int s = 128; s > 0; s >>= 1) { if (tid < s) sred[tid] = fmaxf(sred[tid], sred[tid + s]); __syncthreads(); }
    float rmax = sred[0]; __syncthreads();
    float sum = 0.f;
    for (int gg = tid; gg < NGO; gg += 256) sum += expf(row[gg] - rmax);
    sred[tid] = sum; __syncthreads();
    for (int s = 128; s > 0; s >>= 1) { if (tid < s) sred[tid] += sred[tid + s]; __syncthreads(); }
    if (tid == 0) rowlog[m] = rmax + logf(sred[0]);
}

// ---------------------------------------------------------------------------
// out[b,g] = max over 34 positions of softmax probability.
// ---------------------------------------------------------------------------
__global__ void k_final(const float* __restrict__ S, const float* __restrict__ rowlog,
                        float* __restrict__ out) {
    int g = blockIdx.x * 256 + threadIdx.x;
    int b = blockIdx.y;
    if (g >= NGO) return;
    float best = 0.f;
    for (int l = 0; l < L4_SZ; ++l) {
        int m = b * L4_SZ + l;
        float v = expf(S[(size_t)m * SSTRIDE + g] - rowlog[m]);
        best = fmaxf(best, v);
    }
    out[(size_t)b * NGO + g] = best;
}

// ---------------------------------------------------------------------------
extern "C" void kernel_launch(void* const* d_in, const int* in_sizes, int n_in,
                              void* d_out, int out_size, void* d_ws, size_t ws_size,
                              hipStream_t stream) {
    const float* embed = (const float*)d_in[0];
    const float* go    = (const float*)d_in[1];
    const float* wred  = (const float*)d_in[2];
    const float* bred  = (const float*)d_in[3];
    const float* w1    = (const float*)d_in[4];
    const float* b1    = (const float*)d_in[5];
    const float* w2    = (const float*)d_in[6];
    const float* b2    = (const float*)d_in[7];
    const float* w3    = (const float*)d_in[8];
    const float* b3    = (const float*)d_in[9];
    const float* mobw  = (const float*)d_in[10];
    const float* mobb  = (const float*)d_in[11];
    float* out = (float*)d_out;

    char* ws = (char*)d_ws;
    size_t off = 0;
    auto alloc = [&](size_t bytes) -> void* {
        void* p = ws + off;
        off += (bytes + 255) & ~(size_t)255;
        return p;
    };
    float*  y1     = (float*) alloc((size_t)B_SZ * L0_SZ * CMID * 4);   // 8 MB
    float*  y2     = (float*) alloc((size_t)B_SZ * L2_SZ * CMID * 4);
    float*  y3     = (float*) alloc((size_t)B_SZ * L3_SZ * CMID * 4);
    float*  y4     = (float*) alloc((size_t)B_SZ * L4_SZ * CMID * 4);
    __bf16* wpack  = (__bf16*)alloc((size_t)7 * 32 * 2 * 32 * 16 * 2);  // 448 KB
    __bf16* pbf    = (__bf16*)alloc((size_t)NPTS * DPAD * 2);
    float*  u2v    = (float*) alloc((size_t)NPTS * 4);
    __bf16* gbf    = (__bf16*)alloc((size_t)NGOPAD * DPAD * 2);         // 4.5 MB
    float*  v2v    = (float*) alloc((size_t)NGOPAD * 4);
    float*  Smat   = (float*) alloc((size_t)NPTS * SSTRIDE * 4);        // 44 MB
    float*  rowlog = (float*) alloc((size_t)NPTS * 4);

    // preprocessing
    k_pack_w<<<dim3(896), dim3(256), 0, stream>>>(wred, wpack);
    k_go<<<dim3(NGOPAD), dim3(256), 0, stream>>>(go, gbf, v2v);

    // conv stack
    k_conv1<<<dim3(512), dim3(256), 0, stream>>>(embed, wpack, bred, y1);
    k_convs<<<dim3((B_SZ * L2_SZ * 32 + 255) / 256), dim3(256), 0, stream>>>(y1, w1, b1, y2, L0_SZ, L2_SZ);
    k_convs<<<dim3((B_SZ * L3_SZ * 32 + 255) / 256), dim3(256), 0, stream>>>(y2, w2, b2, y3, L2_SZ, L3_SZ);
    k_convs<<<dim3((B_SZ * L4_SZ * 32 + 255) / 256), dim3(256), 0, stream>>>(y3, w3, b3, y4, L3_SZ, L4_SZ);

    // hyperbolic projection
    k_hyper<<<dim3(NPTS), dim3(256), 0, stream>>>(y4, mobw, mobb, pbf, u2v);

    // distances + softmax + max
    k_dist<<<dim3(NPTS / 16, NGOPAD / 128), dim3(256), 0, stream>>>(pbf, gbf, u2v, v2v, Smat);
    k_rowstats<<<dim3(NPTS), dim3(256), 0, stream>>>(Smat, rowlog);
    k_final<<<dim3((NGO + 255) / 256, B_SZ), dim3(256), 0, stream>>>(Smat, rowlog, out);
}
